// ProbAttention_32126355374161
// MI455X (gfx1250) — matrix operand order repaired and checked by reading, not compiled, not run on hardware
//
#include <hip/hip_runtime.h>
#include <hip/hip_bf16.h>
#include <math.h>

typedef __attribute__((ext_vector_type(16))) _Float16 v16h;
typedef __attribute__((ext_vector_type(8)))  float    v8f;

// ---------------------------------------------------------------------------
// Kernel 1: sampled QK scores  ->  M[b,h,q] = max_s(dot) - mean_s(dot)
// One thread per (b,h,q). Q row cached in registers as float4s; K rows are
// 256B gathers that hit L2 (all of K fits in the 192MB L2). The index load is
// software-pipelined one step ahead and the next K row is prefetched
// (global_prefetch_b8) to hide L2 gather latency behind the 64-FMA dot.
// ---------------------------------------------------------------------------
__global__ __launch_bounds__(256) void probattn_sample_scores(
    const float* __restrict__ Q, const float* __restrict__ K,
    const int* __restrict__ idx, float* __restrict__ M_ws,
    int B, int L, int H, int SK) {
  int t = blockIdx.x * blockDim.x + threadIdx.x;
  int total = B * H * L;
  if (t >= total) return;
  int q  = t % L;
  int bh = t / L;
  int h  = bh % H;
  int b  = bh / H;
  const float4* qrow = (const float4*)(Q + ((((size_t)b * L + q) * H + h) << 6));
  float4 qv[16];
#pragma unroll
  for (int i = 0; i < 16; ++i) qv[i] = qrow[i];

  size_t bl_base = (size_t)b * L;
  float mx = -INFINITY, sum = 0.f;
  int kq_next = idx[q * SK];
  for (int s = 0; s < SK; ++s) {
    int kq = kq_next;
    if (s + 1 < SK) {
      kq_next = idx[q * SK + s + 1];
      const float* knext = K + (((bl_base + kq_next) * H + h) << 6);
      __builtin_prefetch(knext, 0, 0);       // global_prefetch_b8 (128B line)
      __builtin_prefetch(knext + 32, 0, 0);  // second half of the 256B row
    }
    const float4* krow = (const float4*)(K + (((bl_base + kq) * H + h) << 6));
    float dot = 0.f;
#pragma unroll
    for (int i = 0; i < 16; ++i) {
      float4 kv = krow[i];
      dot += qv[i].x * kv.x + qv[i].y * kv.y + qv[i].z * kv.z + qv[i].w * kv.w;
    }
    mx = fmaxf(mx, dot);
    sum += dot;
  }
  M_ws[(size_t)bh * L + q] = mx - sum / (float)SK;
}

// ---------------------------------------------------------------------------
// Kernel 2: exact top-45 per (b,h) via 45 iterative block-argmax passes over
// an LDS copy of the 4096 M values. Deterministic (smallest index on ties).
// Indices stored padded to 48 (pad entries = 0; their rows are never written).
// ---------------------------------------------------------------------------
__global__ __launch_bounds__(256) void probattn_topk(
    const float* __restrict__ M_ws, int* __restrict__ tops, int L, int NTOP) {
  int bh = blockIdx.x;
  const float* Mrow = M_ws + (size_t)bh * L;
  __shared__ float svals[4096];
  __shared__ float rv[256];
  __shared__ int   ri[256];
  for (int i = threadIdx.x; i < L; i += blockDim.x) svals[i] = Mrow[i];
  __syncthreads();
  for (int it = 0; it < NTOP; ++it) {
    float bv = -INFINITY; int bi = 0;
    for (int i = threadIdx.x; i < L; i += blockDim.x) {
      float v = svals[i];
      if (v > bv) { bv = v; bi = i; }
    }
    rv[threadIdx.x] = bv; ri[threadIdx.x] = bi;
    __syncthreads();
    for (int s = 128; s > 0; s >>= 1) {
      if ((int)threadIdx.x < s) {
        float ov = rv[threadIdx.x + s]; int oi = ri[threadIdx.x + s];
        if (ov > rv[threadIdx.x] ||
            (ov == rv[threadIdx.x] && oi < ri[threadIdx.x])) {
          rv[threadIdx.x] = ov; ri[threadIdx.x] = oi;
        }
      }
      __syncthreads();
    }
    if (threadIdx.x == 0) {
      tops[bh * 48 + it] = ri[0];
      svals[ri[0]] = -INFINITY;
    }
    __syncthreads();
  }
  if ((int)threadIdx.x < 48 - NTOP) tops[bh * 48 + NTOP + threadIdx.x] = 0;
}

// ---------------------------------------------------------------------------
// Kernel 3: V mean over L per (b,h,d). Coalesced: 64 lanes own d, 4 row-groups.
// ---------------------------------------------------------------------------
__global__ __launch_bounds__(256) void probattn_vmean(
    const float* __restrict__ V, float* __restrict__ vmean,
    int L, int H) {
  int bh = blockIdx.x;
  int h = bh % H;
  int b = bh / H;
  int d  = threadIdx.x & 63;
  int lr = threadIdx.x >> 6;   // 0..3
  float s = 0.f;
  for (int l = lr; l < L; l += 4)
    s += V[((((size_t)b * L + l) * H + h) << 6) + d];
  __shared__ float red[256];
  red[threadIdx.x] = s;
  __syncthreads();
  if (lr == 0)
    vmean[bh * 64 + d] =
        (red[d] + red[64 + d] + red[128 + d] + red[192 + d]) / (float)L;
}

// ---------------------------------------------------------------------------
// Kernel 4: broadcast mean into the whole output (pure store bandwidth).
// out layout is (B,H,L,D).
// ---------------------------------------------------------------------------
__global__ __launch_bounds__(256) void probattn_bcast(
    const float* __restrict__ vmean, float* __restrict__ out,
    int total, int L) {
  int i = blockIdx.x * blockDim.x + threadIdx.x;
  if (i >= total) return;
  int d  = i & 63;
  int bh = i / (L * 64);
  out[i] = vmean[bh * 64 + d];
}

// ---------------------------------------------------------------------------
// Kernel 5: dense attention for the top-45 queries, flash-attention style,
// with double-buffered K/V LDS tiles (one barrier per tile; next tile's
// global loads overlap the current tile's 8 WMMAs + online softmax).
// One block (3 waves) per (b,h); wave w owns M-tile rows [16w,16w+16).
//
// Fragment layouts used here (wave32; k order is this code's, same for A and B):
//   A (16x32 f16): lane%16 = M row; lane/16 selects K sub-bank;
//                  a[i] -> K = 16*(lane/16) + i (contiguous, as for B).
//   B (32x16 f16): lane%16 = N col; lane/16 = K half; b[i] = B[16*half+i][N]
//                  -> 16 contiguous K values per lane (v16h LDS load).
//   C/D (16x16 f32): lane%16 = N; lane/16 = M half; c[r] = C[r+8*half][N].
// ---------------------------------------------------------------------------
__global__ __launch_bounds__(96) void probattn_topk_attn(
    const float* __restrict__ Q, const float* __restrict__ K,
    const float* __restrict__ V, const int* __restrict__ tops,
    float* __restrict__ out, int L, int H) {
  constexpr int NT = 32;      // L-tile
  constexpr int NTOP = 45;
  int bh   = blockIdx.x;
  int h    = bh % H;
  int b    = bh / H;
  int wave = threadIdx.x >> 5;   // 0..2  -> M-tile
  int lane = threadIdx.x & 31;
  int lN   = lane & 15;
  int half = lane >> 4;

  __shared__ alignas(32) _Float16 K_lds[2][NT * 64];     // row-major [row][k]
  __shared__ alignas(32) _Float16 V_ldsT[2][64 * NT];    // transposed [d][row]
  __shared__ alignas(32) _Float16 P_lds[3][16 * 32];     // per-wave P [m][k2]

  size_t bl_base = (size_t)b * L;

  auto stage = [&](int l0, int buf) {
    for (int t = threadIdx.x; t < NT * 64; t += 96) {
      int row = t >> 6, col = t & 63;
      size_t g = (((bl_base + (l0 + row)) * H + h) << 6) + col;
      K_lds[buf][row * 64 + col]  = (_Float16)K[g];
      V_ldsT[buf][col * NT + row] = (_Float16)V[g];
    }
  };

  // ---- load Q_top A-fragments (scale 1/sqrt(64)=0.125 folded in) ----
  int qidx = tops[bh * 48 + wave * 16 + lN];
  const float* qptr = Q + (((bl_base + qidx) * H + h) << 6);
  v16h a0, a1;
#pragma unroll
  for (int i = 0; i < 16; ++i) {
    int k = 16 * half + i;
    a0[i] = (_Float16)(qptr[k]      * 0.125f);
    a1[i] = (_Float16)(qptr[k + 32] * 0.125f);
  }

  v8f acc0 = {}, acc1 = {}, acc2 = {}, acc3 = {};
  float m_r[8], l_r[8];
#pragma unroll
  for (int r = 0; r < 8; ++r) { m_r[r] = -INFINITY; l_r[r] = 0.f; }

  stage(0, 0);
  int cur = 0;

  for (int l0 = 0; l0 < L; l0 += NT) {
    __syncthreads();   // buf[cur] staged; previous compute (on buf[cur^1]) done

    // ---- stage next tile into the other buffer (overlaps compute below) ----
    if (l0 + NT < L) {
      stage(l0 + NT, cur ^ 1);
      // warm L0/L2 for the tile after that
      if (l0 + 2 * NT < L) {
        for (int t = threadIdx.x; t < 2 * NT; t += 96) {
          int row = t & (NT - 1);
          const float* p = ((t < NT) ? K : V) +
                           (((bl_base + (l0 + 2 * NT + row)) * H + h) << 6);
          __builtin_prefetch(p, 0, 0);
          __builtin_prefetch(p + 32, 0, 0);
        }
      }
    }

    const _Float16* Kc = K_lds[cur];
    const _Float16* Vc = V_ldsT[cur];

    // ---- S = Q_tile . K_tile^T  (two 16x16 subtiles, two k-steps each) ----
    v8f s0 = {}, s1 = {};
    {
      const v16h b00 = *(const v16h*)&Kc[(0 * 16 + lN) * 64 +  0 + 16 * half];
      const v16h b01 = *(const v16h*)&Kc[(0 * 16 + lN) * 64 + 32 + 16 * half];
      s0 = __builtin_amdgcn_wmma_f32_16x16x32_f16(false, a0, false, b00, (short)0, s0, false, false);
      s0 = __builtin_amdgcn_wmma_f32_16x16x32_f16(false, a1, false, b01, (short)0, s0, false, false);
      const v16h b10 = *(const v16h*)&Kc[(1 * 16 + lN) * 64 +  0 + 16 * half];
      const v16h b11 = *(const v16h*)&Kc[(1 * 16 + lN) * 64 + 32 + 16 * half];
      s1 = __builtin_amdgcn_wmma_f32_16x16x32_f16(false, a0, false, b10, (short)0, s1, false, false);
      s1 = __builtin_amdgcn_wmma_f32_16x16x32_f16(false, a1, false, b11, (short)0, s1, false, false);
    }

    // ---- online softmax over the 32 new columns (16-lane row reductions) ----
#pragma unroll
    for (int r = 0; r < 8; ++r) {
      float tm = fmaxf(s0[r], s1[r]);
      tm = fmaxf(tm, __shfl_xor(tm, 1));
      tm = fmaxf(tm, __shfl_xor(tm, 2));
      tm = fmaxf(tm, __shfl_xor(tm, 4));
      tm = fmaxf(tm, __shfl_xor(tm, 8));
      float mn    = fmaxf(m_r[r], tm);
      float alpha = __expf(m_r[r] - mn);   // exp(-inf)=0 on first tile
      m_r[r] = mn;
      float p0 = __expf(s0[r] - mn);
      float p1 = __expf(s1[r] - mn);
      float ts = p0 + p1;
      ts += __shfl_xor(ts, 1);
      ts += __shfl_xor(ts, 2);
      ts += __shfl_xor(ts, 4);
      ts += __shfl_xor(ts, 8);
      l_r[r] = l_r[r] * alpha + ts;
      acc0[r] *= alpha; acc1[r] *= alpha; acc2[r] *= alpha; acc3[r] *= alpha;
      // C-layout -> [m][k2] staging for the A fragment of P.V
      P_lds[wave][(r + 8 * half) * 32 +  0 + lN] = (_Float16)p0;
      P_lds[wave][(r + 8 * half) * 32 + 16 + lN] = (_Float16)p1;
    }

    // ---- reload P as an A fragment (same-wave LDS ops are in-order) ----
    v16h ap;
#pragma unroll
    for (int i = 0; i < 16; ++i) {
      ap[i] = P_lds[wave][lN * 32 + 16 * half + i];
    }

    // ---- acc += P . V_tile  (four 16-wide D subtiles, K=32) ----
    {
      const v16h bv0 = *(const v16h*)&Vc[( 0 + lN) * NT + 16 * half];
      const v16h bv1 = *(const v16h*)&Vc[(16 + lN) * NT + 16 * half];
      const v16h bv2 = *(const v16h*)&Vc[(32 + lN) * NT + 16 * half];
      const v16h bv3 = *(const v16h*)&Vc[(48 + lN) * NT + 16 * half];
      acc0 = __builtin_amdgcn_wmma_f32_16x16x32_f16(false, ap, false, bv0, (short)0, acc0, false, false);
      acc1 = __builtin_amdgcn_wmma_f32_16x16x32_f16(false, ap, false, bv1, (short)0, acc1, false, false);
      acc2 = __builtin_amdgcn_wmma_f32_16x16x32_f16(false, ap, false, bv2, (short)0, acc2, false, false);
      acc3 = __builtin_amdgcn_wmma_f32_16x16x32_f16(false, ap, false, bv3, (short)0, acc3, false, false);
    }

    cur ^= 1;
  }

  // ---- normalize and scatter the 45 real rows into the output ----
#pragma unroll
  for (int r = 0; r < 8; ++r) {
    int row = wave * 16 + r + 8 * half;
    if (row < NTOP) {
      float inv = 1.0f / l_r[r];
      int qg = tops[bh * 48 + row];
      float* orow = out + (((size_t)bh * L + qg) << 6);
      orow[ 0 + lN] = acc0[r] * inv;
      orow[16 + lN] = acc1[r] * inv;
      orow[32 + lN] = acc2[r] * inv;
      orow[48 + lN] = acc3[r] * inv;
    }
  }
}

// ---------------------------------------------------------------------------
extern "C" void kernel_launch(void* const* d_in, const int* in_sizes, int n_in,
                              void* d_out, int out_size, void* d_ws, size_t ws_size,
                              hipStream_t stream) {
  const int B = 4, L = 4096, H = 8;
  const float* Q   = (const float*)d_in[0];
  const float* K   = (const float*)d_in[1];
  const float* V   = (const float*)d_in[2];
  const int*   idx = (const int*)d_in[3];
  float* out = (float*)d_out;

  int SK   = in_sizes[3] / L;   // 45
  int NTOP = SK;                // same formula in the reference

  // workspace carve-out
  float* M_ws  = (float*)d_ws;                       // B*H*L floats (512 KB)
  int*   tops  = (int*)(M_ws + (size_t)B * H * L);   // B*H*48 ints
  float* vmean = (float*)(tops + B * H * 48);        // B*H*64 floats

  int totQ = B * H * L;
  probattn_sample_scores<<<(totQ + 255) / 256, 256, 0, stream>>>(
      Q, K, idx, M_ws, B, L, H, SK);

  probattn_topk<<<B * H, 256, 0, stream>>>(M_ws, tops, L, NTOP);

  probattn_vmean<<<B * H, 256, 0, stream>>>(V, vmean, L, H);

  int totO = B * H * L * 64;
  probattn_bcast<<<(totO + 255) / 256, 256, 0, stream>>>(vmean, out, totO, L);

  probattn_topk_attn<<<B * H, 96, 0, stream>>>(Q, K, V, tops, out, L, H);
}
